// Attention_46196668236091
// MI455X (gfx1250) — matrix operand order repaired and checked by reading, not compiled, not run on hardware
//
#include <hip/hip_runtime.h>
#include <hip/hip_bf16.h>
#include <stdint.h>

typedef __attribute__((ext_vector_type(16))) __bf16 v16bf;
typedef __attribute__((ext_vector_type(8)))  float  v8f;

#define BATCH 32
#define IDF   256
#define CDF   256
#define HXW   4096
#define LCTX  128
#define PB    32     // p positions per workgroup (2 WMMA M-tiles)
#define NTHR  256    // 8 waves

struct U16x16 { uint4 a, b; };

__device__ __forceinline__ unsigned short f2bf(float f) {
    union { float f; unsigned u; } v; v.f = f;
    unsigned u = v.u;
    return (unsigned short)((u + 0x7FFFu + ((u >> 16) & 1u)) >> 16);
}

// A-fragment: A[m][k], 16x32 bf16, row-major with row stride S (elements).
// lane<16 : row=lane,    K = {0..7, 16..23}
// lane>=16: row=lane-16, K = {8..15, 24..31}
__device__ __forceinline__ v16bf load_a_frag(const unsigned short* base, int S,
                                             int k0, int lane) {
    int m  = lane & 15;
    int kb = (lane < 16) ? 0 : 8;
    const unsigned short* p = base + m * S + k0 + kb;
    U16x16 t;
    t.a = *(const uint4*)(p);       // K = kb .. kb+7
    t.b = *(const uint4*)(p + 16);  // K = kb+16 .. kb+23
    return __builtin_bit_cast(v16bf, t);
}

// B-fragment: B[k][n], 32x16 bf16, supplied TRANSPOSED as bt[n][k], stride S.
// lane<16 : col=n0+lane,    K = {0..7, 16..23}
// lane>=16: col=n0+lane-16, K = {8..15, 24..31}
__device__ __forceinline__ v16bf load_b_frag(const unsigned short* bt, int n0,
                                             int S, int k0, int lane) {
    int n  = lane & 15;
    int ko = (lane < 16) ? 0 : 8;
    const unsigned short* p = bt + (size_t)(n0 + n) * S + k0 + ko;
    U16x16 t;
    t.a = *(const uint4*)(p);
    t.b = *(const uint4*)(p + 16);
    return __builtin_bit_cast(v16bf, t);
}

// ---------------------------------------------------------------------------
// Kernel 1: source[b] = W x context[b]; write bf16 in both [i][l] and [l][i]
// ---------------------------------------------------------------------------
__global__ __launch_bounds__(128) void proj_kernel(
    const float* __restrict__ W, const float* __restrict__ context,
    unsigned short* __restrict__ src_il, unsigned short* __restrict__ src_li) {
    __shared__ __align__(16) float Wsh[16][CDF];
    const int b  = blockIdx.y;
    const int i0 = blockIdx.x * 16;
    const int tid = threadIdx.x;
    for (int idx = tid; idx < 16 * CDF; idx += 128)
        Wsh[idx >> 8][idx & 255] = W[(size_t)(i0 + (idx >> 8)) * CDF + (idx & 255)];
    __syncthreads();
    const int l = tid;  // 128 threads == LCTX
    const float* ctx = context + (size_t)b * CDF * LCTX;
    float acc[16];
#pragma unroll
    for (int ii = 0; ii < 16; ++ii) acc[ii] = 0.f;
    for (int c = 0; c < CDF; ++c) {
        float cv = ctx[(size_t)c * LCTX + l];
#pragma unroll
        for (int ii = 0; ii < 16; ++ii) acc[ii] = fmaf(Wsh[ii][c], cv, acc[ii]);
    }
#pragma unroll
    for (int ii = 0; ii < 16; ++ii) {
        unsigned short h = f2bf(acc[ii]);
        src_il[((size_t)b * IDF + i0 + ii) * LCTX + l] = h;   // [i][l]
        src_li[((size_t)b * LCTX + l) * IDF + i0 + ii] = h;   // [l][i]
    }
}

// ---------------------------------------------------------------------------
// Kernel 2: per (b, 32-p block): QK^T -> masked softmax -> PV, all via WMMA
// ---------------------------------------------------------------------------
__global__ __launch_bounds__(NTHR) void attn_kernel(
    const float* __restrict__ x,
    const unsigned short* __restrict__ src_il,
    const unsigned short* __restrict__ src_li,
    const unsigned char* __restrict__ mask,
    float* __restrict__ out_result,
    float* __restrict__ out_attn) {
    __shared__ __align__(16) unsigned short xtile[PB][IDF];   // 16 KB bf16 x^T
    __shared__ __align__(16) float          scores[PB][LCTX]; // 16 KB
    __shared__ __align__(16) unsigned short probs[PB][LCTX];  //  8 KB bf16
    __shared__               float          red[PB][8];       //  1 KB
    __shared__ __align__(16) float          otile[PB][IDF];   // 32 KB

    const int b    = blockIdx.y;
    const int p0   = blockIdx.x * PB;
    const int tid  = threadIdx.x;
    const int wave = tid >> 5;
    const int lane = tid & 31;

    // ---- stage x tile (transpose + bf16): xtile[p][i], 128B segments ----
    {
        const int p = tid & 31;
        for (int i = tid >> 5; i < IDF; i += 8)
            xtile[p][i] = f2bf(x[((size_t)b * IDF + i) * HXW + p0 + p]);
    }
    __syncthreads();

    // ---- QK^T: D[m=p][n=l], K=i; wave -> (m-tile, 2 l-tiles) ------------
    const unsigned short* bt_li = src_li + (size_t)b * LCTX * IDF;  // [l][i]
    {
        const int mt = wave >> 2;          // 0..1 : p subtile
        const int nq = (wave & 3) * 32;    // l offset (two 16-wide tiles)
        v8f acc0 = {}, acc1 = {};
        for (int k0 = 0; k0 < IDF; k0 += 32) {
            v16bf af = load_a_frag(&xtile[mt * 16][0], IDF, k0, lane);
            v16bf b0 = load_b_frag(bt_li, nq,      IDF, k0, lane);
            v16bf b1 = load_b_frag(bt_li, nq + 16, IDF, k0, lane);
            acc0 = __builtin_amdgcn_wmma_f32_16x16x32_bf16(false, af, false, b0,
                                                           (short)0, acc0, false, false);
            acc1 = __builtin_amdgcn_wmma_f32_16x16x32_bf16(false, af, false, b1,
                                                           (short)0, acc1, false, false);
        }
        const int n  = lane & 15;
        const int mb = mt * 16 + ((lane < 16) ? 0 : 8);
#pragma unroll
        for (int r = 0; r < 8; ++r) {
            scores[mb + r][nq + n]      = acc0[r];
            scores[mb + r][nq + 16 + n] = acc1[r];
        }
    }
    __syncthreads();

    // ---- masked softmax over L (torch tile quirk: mask row = idx % B) ---
    {
        const int p   = tid >> 3;   // 0..31
        const int seg = tid & 7;    // 16 l's each
        const unsigned char* mrow =
            mask + (size_t)((b * HXW + p0 + p) % BATCH) * LCTX;
        float ev[16];
        float mx = -INFINITY;
#pragma unroll
        for (int j = 0; j < 16; ++j) {
            const int l = seg * 16 + j;
            float s = mrow[l] ? -INFINITY : scores[p][l];
            ev[j] = s;
            mx = fmaxf(mx, s);
        }
        red[p][seg] = mx;
        __syncthreads();
        float rmax = red[p][0];
#pragma unroll
        for (int j = 1; j < 8; ++j) rmax = fmaxf(rmax, red[p][j]);
        __syncthreads();
        float ssum = 0.f;
#pragma unroll
        for (int j = 0; j < 16; ++j) {
            float e = (ev[j] == -INFINITY) ? 0.f : __expf(ev[j] - rmax);
            ev[j] = e;
            ssum += e;
        }
        red[p][seg] = ssum;
        __syncthreads();
        float tot = 0.f;
#pragma unroll
        for (int j = 0; j < 8; ++j) tot += red[p][j];
        const float inv = 1.0f / tot;
#pragma unroll
        for (int j = 0; j < 16; ++j) {
            const int l = seg * 16 + j;
            const float pr = ev[j] * inv;
            scores[p][l] = pr;        // f32 for attn output
            probs[p][l]  = f2bf(pr);  // bf16 for PV
        }
    }
    __syncthreads();

    // ---- attn_t writeout (128B coalesced over p) ------------------------
    {
        float* ap = out_attn + (size_t)b * LCTX * HXW + p0;
        const int pp = tid & 31;
        for (int l = tid >> 5; l < LCTX; l += 8)
            ap[(size_t)l * HXW + pp] = scores[pp][l];
    }

    // ---- PV: D[m=p][n=i], K=l; wave -> (m-tile, 4 i-tiles) --------------
    const unsigned short* bt_il = src_il + (size_t)b * IDF * LCTX;  // [i][l]
    {
        const int mt = wave >> 2;         // 0..1
        const int nb = (wave & 3) * 4;    // first of 4 i-tiles
        v8f oacc[4];
#pragma unroll
        for (int t = 0; t < 4; ++t) oacc[t] = (v8f){};
        for (int k0 = 0; k0 < LCTX; k0 += 32) {
            v16bf af = load_a_frag(&probs[mt * 16][0], LCTX, k0, lane);
#pragma unroll
            for (int t = 0; t < 4; ++t) {
                v16bf bf = load_b_frag(bt_il, (nb + t) * 16, LCTX, k0, lane);
                oacc[t] = __builtin_amdgcn_wmma_f32_16x16x32_bf16(false, af, false, bf,
                                                                  (short)0, oacc[t],
                                                                  false, false);
            }
        }
        const int n  = lane & 15;
        const int mb = mt * 16 + ((lane < 16) ? 0 : 8);
#pragma unroll
        for (int t = 0; t < 4; ++t)
#pragma unroll
            for (int r = 0; r < 8; ++r)
                otile[mb + r][(nb + t) * 16 + n] = oacc[t][r];
    }
    __syncthreads();

    // ---- result writeout (128B coalesced over p) ------------------------
    {
        float* rp = out_result + (size_t)b * IDF * HXW + p0;
        const int pp = tid & 31;
        for (int i = tid >> 5; i < IDF; i += 8)
            rp[(size_t)i * HXW + pp] = otile[pp][i];
    }
}

// ---------------------------------------------------------------------------
extern "C" void kernel_launch(void* const* d_in, const int* in_sizes, int n_in,
                              void* d_out, int out_size, void* d_ws, size_t ws_size,
                              hipStream_t stream) {
    const float* x       = (const float*)d_in[0];
    const float* context = (const float*)d_in[1];
    const float* W       = (const float*)d_in[2];
    const unsigned char* mask = (const unsigned char*)d_in[3];  // jax bool = 1 byte

    float* out_result = (float*)d_out;                               // [B,idf,h,w]
    float* out_attn   = out_result + (size_t)BATCH * IDF * HXW;      // [B,L,h,w]

    unsigned short* src_il = (unsigned short*)d_ws;                  // 2 MB
    unsigned short* src_li = src_il + (size_t)BATCH * IDF * LCTX;    // 2 MB

    dim3 pgrid(IDF / 16, BATCH);
    proj_kernel<<<pgrid, 128, 0, stream>>>(W, context, src_il, src_li);

    dim3 agrid(HXW / PB, BATCH);
    attn_kernel<<<agrid, NTHR, 0, stream>>>(x, src_il, src_li, mask,
                                            out_result, out_attn);
}
